// Attention_58007828300516
// MI455X (gfx1250) — matrix operand order repaired and checked by reading, not compiled, not run on hardware
//
#include <hip/hip_runtime.h>

typedef __bf16 v16bf __attribute__((ext_vector_type(16)));
typedef float  v8f   __attribute__((ext_vector_type(8)));

#define TPB    256
#define WAVES  8
#define BK     32
#define DD     128
#define KPITCH 136   // bf16 elems per K row in LDS  (272B = 68 dwords, conflict-free)
#define VPITCH 34    // bf16 elems per V^T row       ( 68B = 17 dwords)
#define PPITCH 34    // bf16 elems per P row

union BV {
  v16bf v;
  uint4 q[2];
  unsigned int u[8];
};
union PK4 {
  __bf16 h[4];
  uint2  u2;
};

__device__ __forceinline__ float rmax16(float x) {
  x = fmaxf(x, __shfl_xor(x, 1, 32));
  x = fmaxf(x, __shfl_xor(x, 2, 32));
  x = fmaxf(x, __shfl_xor(x, 4, 32));
  x = fmaxf(x, __shfl_xor(x, 8, 32));
  return x;
}
__device__ __forceinline__ float rsum16(float x) {
  x += __shfl_xor(x, 1, 32);
  x += __shfl_xor(x, 2, 32);
  x += __shfl_xor(x, 4, 32);
  x += __shfl_xor(x, 8, 32);
  return x;
}

__device__ __forceinline__ v8f wmma_bf16(v16bf a, v16bf b, v8f c) {
  return __builtin_amdgcn_wmma_f32_16x16x32_bf16(false, a, false, b, (short)0, c,
                                                 false, false);
}

// NKC > 0: compile-time NK (immediate-offset mask addressing). NKC == 0: runtime.
template <int NKC>
__global__ void __launch_bounds__(TPB)
fa_mask_wmma(const float* __restrict__ Qg, const float* __restrict__ Kg,
             const float* __restrict__ Vg, const float* __restrict__ Mg,
             float* __restrict__ Og, int NQ, int NKrt)
{
  const int NK = (NKC > 0) ? NKC : NKrt;

  __shared__ __align__(16) __bf16 kb_s[2][32 * KPITCH];     // K block, row-major bf16
  __shared__ __align__(16) __bf16 vt_s[2][DD * VPITCH];     // V block, transposed bf16
  __shared__ __align__(16) __bf16 p_s[WAVES][16 * PPITCH];  // per-wave P scratch

  const int tid  = (int)threadIdx.x;
  const int b    = blockIdx.y;
  const int wave = tid >> 5;
  const int lane = tid & 31;
  const int n    = lane & 15;   // N-dim lane (cols of C/D)
  const int h    = lane >> 4;   // lane half
  const int qbase = blockIdx.x * (WAVES * 16) + wave * 16;

  const float* Qb = Qg + (size_t)b * NQ * DD;
  const float* Kb = Kg + (size_t)b * NK * DD;
  const float* Vb = Vg + (size_t)b * NK * DD;
  const float* Mb = Mg + (size_t)b * NQ * (size_t)NK;
  float*       Ob = Og + (size_t)b * NQ * DD;

  // uniform mask base (SGPR) + loop-invariant per-lane offset; row steps are immediates
  const float* mbase = Mb + (size_t)qbase * NK;
  const int    moff  = (8 * h) * NK + n;

  // ---- Q tile -> resident A-layout bf16 regs, pre-scaled by 1/sqrt(D) ----
  v16bf a[4];
  {
    const float SC = 0.08838834764831845f;  // 1/sqrt(128)
    const float* qrow = Qb + (size_t)(qbase + n) * DD;
    #pragma unroll
    for (int c = 0; c < 4; ++c) {
      const int d0 = 32 * c + 8 * h;
      float4 r0 = *(const float4*)(qrow + d0);
      float4 r1 = *(const float4*)(qrow + d0 + 4);
      float4 r2 = *(const float4*)(qrow + d0 + 16);
      float4 r3 = *(const float4*)(qrow + d0 + 20);
      v16bf t;
      t[0]  = (__bf16)(r0.x * SC); t[1]  = (__bf16)(r0.y * SC);
      t[2]  = (__bf16)(r0.z * SC); t[3]  = (__bf16)(r0.w * SC);
      t[4]  = (__bf16)(r1.x * SC); t[5]  = (__bf16)(r1.y * SC);
      t[6]  = (__bf16)(r1.z * SC); t[7]  = (__bf16)(r1.w * SC);
      t[8]  = (__bf16)(r2.x * SC); t[9]  = (__bf16)(r2.y * SC);
      t[10] = (__bf16)(r2.z * SC); t[11] = (__bf16)(r2.w * SC);
      t[12] = (__bf16)(r3.x * SC); t[13] = (__bf16)(r3.y * SC);
      t[14] = (__bf16)(r3.z * SC); t[15] = (__bf16)(r3.w * SC);
      a[c] = t;
    }
  }

  // ---- stage block 0 into LDS buffer 0 ----
  #pragma unroll
  for (int i = 0; i < 4; ++i) {
    const int linear = tid + TPB * i;
    const int kr = linear >> 5;
    const int dq = (linear & 31) << 2;
    const size_t goff = (size_t)kr * DD + dq;
    float4 kv = *(const float4*)(Kb + goff);
    float4 vv = *(const float4*)(Vb + goff);
    PK4 pk;
    pk.h[0] = (__bf16)kv.x; pk.h[1] = (__bf16)kv.y;
    pk.h[2] = (__bf16)kv.z; pk.h[3] = (__bf16)kv.w;
    *(uint2*)&kb_s[0][kr * KPITCH + dq] = pk.u2;
    vt_s[0][(dq + 0) * VPITCH + kr] = (__bf16)vv.x;
    vt_s[0][(dq + 1) * VPITCH + kr] = (__bf16)vv.y;
    vt_s[0][(dq + 2) * VPITCH + kr] = (__bf16)vv.z;
    vt_s[0][(dq + 3) * VPITCH + kr] = (__bf16)vv.w;
  }

  // accumulators + running stats (row = j + 8h, replicated across 16-lane halves)
  v8f acc[8];
  v8f Mx, Lr, Zr;
  #pragma unroll
  for (int c = 0; c < 8; ++c) {
    #pragma unroll
    for (int j = 0; j < 8; ++j) acc[c][j] = 0.0f;
  }
  #pragma unroll
  for (int j = 0; j < 8; ++j) { Mx[j] = -__builtin_inff(); Lr[j] = 0.0f; Zr[j] = 0.0f; }

  for (int kb = 0; kb < NK; kb += BK) {
    const int ibuf = (kb >> 5) & 1;
    __syncthreads();  // buffer ibuf fully staged; buffer ibuf^1 free to overwrite

    // ---- issue NEXT block's global loads (wait lands after the compute) ----
    const bool hn = (kb + BK) < NK;
    float4 kv[4], vv[4];
    if (hn) {
      #pragma unroll
      for (int i = 0; i < 4; ++i) {
        const int linear = tid + TPB * i;
        const int kr = linear >> 5;
        const int dq = (linear & 31) << 2;
        const size_t goff = (size_t)(kb + BK + kr) * DD + dq;
        kv[i] = *(const float4*)(Kb + goff);
        vv[i] = *(const float4*)(Vb + goff);
      }
    }

    // ---- mask tiles: uniform base + immediate row offsets, non-temporal ----
    float mk0[8], mk1[8];
    {
      const float* mp = mbase + kb + moff;
      #pragma unroll
      for (int j = 0; j < 8; ++j) {
        mk0[j] = __builtin_nontemporal_load(mp + j * NK);
        mk1[j] = __builtin_nontemporal_load(mp + j * NK + 16);
      }
    }

    // ---- S = Q·K^T, double-buffered B operands from LDS ----
    const __bf16* kbb = &kb_s[ibuf][n * KPITCH + 8 * h];
    auto ldb = [&](int u, int c) {
      BV r;
      const __bf16* p = kbb + u * (16 * KPITCH) + 32 * c;
      r.q[0] = *(const uint4*)(p);
      r.q[1] = *(const uint4*)(p + 16);
      return r;
    };
    v8f s0, s1;
    #pragma unroll
    for (int j = 0; j < 8; ++j) { s0[j] = 0.0f; s1[j] = 0.0f; }
    {
      BV bA = ldb(0, 0);
      BV bB = ldb(0, 1);
      s0 = wmma_bf16(a[0], bA.v, s0); bA = ldb(0, 2);
      s0 = wmma_bf16(a[1], bB.v, s0); bB = ldb(0, 3);
      s0 = wmma_bf16(a[2], bA.v, s0); bA = ldb(1, 0);
      s0 = wmma_bf16(a[3], bB.v, s0); bB = ldb(1, 1);
      s1 = wmma_bf16(a[0], bA.v, s1); bA = ldb(1, 2);
      s1 = wmma_bf16(a[1], bB.v, s1); bB = ldb(1, 3);
      s1 = wmma_bf16(a[2], bA.v, s1);
      s1 = wmma_bf16(a[3], bB.v, s1);
    }

    // ---- online masked-softmax update ----
    #pragma unroll
    for (int j = 0; j < 8; ++j) {
      float bm   = rmax16(fmaxf(s0[j], s1[j]));
      float newM = fmaxf(Mx[j], bm);
      float sc   = __expf(Mx[j] - newM);
      Mx[j] = newM;
      float p0 = __expf(s0[j] - newM);
      float p1 = __expf(s1[j] - newM);
      Zr[j] = Zr[j] * sc + rsum16(p0 + p1);
      float q0 = p0 * mk0[j];
      float q1 = p1 * mk1[j];
      Lr[j] = Lr[j] * sc + rsum16(q0 + q1);
      #pragma unroll
      for (int c = 0; c < 8; ++c) acc[c][j] *= sc;
      p_s[wave][(j + 8 * h) * PPITCH + n]      = (__bf16)q0;
      p_s[wave][(j + 8 * h) * PPITCH + 16 + n] = (__bf16)q1;
    }
    __builtin_amdgcn_wave_barrier();  // per-wave LDS RAW ordering

    // ---- P (16x32) C-layout -> A-layout via LDS ----
    BV ap;
    #pragma unroll
    for (int j = 0; j < 4; ++j)
      ap.u[j] = *(const unsigned int*)&p_s[wave][n * PPITCH + 8 * h + 2 * j];
    #pragma unroll
    for (int j = 4; j < 8; ++j)
      ap.u[j] = *(const unsigned int*)&p_s[wave][n * PPITCH + 16 + 8 * h + 2 * (j - 4)];

    // ---- O += P·V, double-buffered B operands from transposed V ----
    const __bf16* vbb = &vt_s[ibuf][n * VPITCH + 8 * h];
    auto ldv = [&](int c) {
      BV r;
      const __bf16* p = vbb + c * (16 * VPITCH);
      #pragma unroll
      for (int j = 0; j < 8; ++j) r.u[j] = *(const unsigned int*)(p + 2 * j + 8 * (j >> 2));
      return r;
    };
    {
      BV bA = ldv(0);
      BV bB = ldv(1);
      acc[0] = wmma_bf16(ap.v, bA.v, acc[0]); bA = ldv(2);
      acc[1] = wmma_bf16(ap.v, bB.v, acc[1]); bB = ldv(3);
      acc[2] = wmma_bf16(ap.v, bA.v, acc[2]); bA = ldv(4);
      acc[3] = wmma_bf16(ap.v, bB.v, acc[3]); bB = ldv(5);
      acc[4] = wmma_bf16(ap.v, bA.v, acc[4]); bA = ldv(6);
      acc[5] = wmma_bf16(ap.v, bB.v, acc[5]); bB = ldv(7);
      acc[6] = wmma_bf16(ap.v, bA.v, acc[6]);
      acc[7] = wmma_bf16(ap.v, bB.v, acc[7]);
    }

    // ---- drain pipeline: convert + store NEXT block into the other LDS buffer ----
    if (hn) {
      #pragma unroll
      for (int i = 0; i < 4; ++i) {
        const int linear = tid + TPB * i;
        const int kr = linear >> 5;
        const int dq = (linear & 31) << 2;
        PK4 pk;
        pk.h[0] = (__bf16)kv[i].x; pk.h[1] = (__bf16)kv[i].y;
        pk.h[2] = (__bf16)kv[i].z; pk.h[3] = (__bf16)kv[i].w;
        *(uint2*)&kb_s[ibuf ^ 1][kr * KPITCH + dq] = pk.u2;
        vt_s[ibuf ^ 1][(dq + 0) * VPITCH + kr] = (__bf16)vv[i].x;
        vt_s[ibuf ^ 1][(dq + 1) * VPITCH + kr] = (__bf16)vv[i].y;
        vt_s[ibuf ^ 1][(dq + 2) * VPITCH + kr] = (__bf16)vv[i].z;
        vt_s[ibuf ^ 1][(dq + 3) * VPITCH + kr] = (__bf16)vv[i].w;
      }
    }
  }

  // ---- epilogue: out = O / (L + eps*Z), coalesced non-temporal stores ----
  const float EPS = 1e-6f;
  #pragma unroll
  for (int j = 0; j < 8; ++j) {
    const float inv = 1.0f / (Lr[j] + EPS * Zr[j]);
    float* orow = Ob + (size_t)(qbase + j + 8 * h) * DD;
    #pragma unroll
    for (int c = 0; c < 8; ++c)
      __builtin_nontemporal_store(acc[c][j] * inv, orow + 16 * c + n);
  }
}

extern "C" void kernel_launch(void* const* d_in, const int* in_sizes, int n_in,
                              void* d_out, int out_size, void* d_ws, size_t ws_size,
                              hipStream_t stream)
{
  const float* Q = (const float*)d_in[0];
  const float* K = (const float*)d_in[1];
  const float* V = (const float*)d_in[2];
  const float* M = (const float*)d_in[3];
  float* O = (float*)d_out;

  const long D = 128;
  const long nq_rows = (long)in_sizes[0] / D;  // N*NQ
  const long nk_rows = (long)in_sizes[1] / D;  // N*NK
  const long me      = (long)in_sizes[3];      // N*NQ*NK
  const long NK = me / nq_rows;
  const long NQ = me / nk_rows;
  const long N  = nq_rows / NQ;

  dim3 grid((unsigned)(NQ / (WAVES * 16)), (unsigned)N);
  if (NK == 2048) {
    fa_mask_wmma<2048><<<grid, TPB, 0, stream>>>(Q, K, V, M, O, (int)NQ, (int)NK);
  } else {
    fa_mask_wmma<0><<<grid, TPB, 0, stream>>>(Q, K, V, M, O, (int)NQ, (int)NK);
  }
}
